// OptimizedGeometricAttention_43052752175156
// MI455X (gfx1250) — matrix operand order repaired and checked by reading, not compiled, not run on hardware
//
#include <hip/hip_runtime.h>
#include <math.h>

// ---------------------------------------------------------------------------
// Geometric attention for MI455X (gfx1250, wave32, WMMA + TDM).
// All three matmuls run on v_wmma_f32_16x16x32_bf16 (f32 accumulate).
// 128x64 block tiles, 8 waves (4x2), 32x32 C-tile per wave; K-step 64 ->
// 8 WMMAs per wave per barrier round. Q/K/V bf16 tiles staged into LDS by
// the Tensor Data Mover (tensor_load_to_lds + s_wait_tensorcnt).
// ---------------------------------------------------------------------------

typedef __attribute__((ext_vector_type(16))) __bf16 v16bf;
typedef __attribute__((ext_vector_type(8)))  float  v8f;
typedef unsigned int u32x4 __attribute__((ext_vector_type(4)));
typedef int          i32x8 __attribute__((ext_vector_type(8)));
typedef int          i32x4 __attribute__((ext_vector_type(4)));

#define BATCH 4
#define SEQ   4096
#define DIM   512
#define NROW  (BATCH * SEQ)     // 16384 rows total
#define THRC  0.01f
#define INV_SQRT_D 0.044194173824159216f   // 512^-0.5

#if defined(__has_builtin)
#if __has_builtin(__builtin_amdgcn_tensor_load_to_lds)
#define GA_HAVE_TDM 1
#endif
#endif
#ifndef GA_HAVE_TDM
#define GA_HAVE_TDM 0
#warning "GA probe: __builtin_amdgcn_tensor_load_to_lds NOT available; using manual LDS staging"
#endif

// ----------------------- Tensor Data Mover descriptor ----------------------
// 2D tile load, element size 2 bytes (bf16). D# layout per ISA 08 §8.3/8.4:
//  group0: [1:0]=count=1, [63:32]=lds_addr, [120:64]=global_addr, [127:126]=2
//  group1: [17:16]=data_size(1->2B), [79:48]=tensor_dim0, [111:80]=tensor_dim1,
//          [127:112]=tile_dim0, [143:128]=tile_dim1, [207:160]=dim0_stride
#if GA_HAVE_TDM
__device__ static inline void tdm_load_2d(unsigned lds_off, const void* gptr,
                                          unsigned tile_d0, unsigned tile_d1,
                                          unsigned tensor_d0, unsigned tensor_d1,
                                          unsigned long long stride0) {
    const unsigned long long ga = (unsigned long long)(size_t)gptr;
    u32x4 g0;
    g0[0] = 1u;                                             // count=1, user D#
    g0[1] = lds_off;                                        // lds_addr
    g0[2] = (unsigned)(ga & 0xffffffffull);                 // global_addr lo
    g0[3] = (unsigned)((ga >> 32) & 0x01ffffffull) | (2u << 30);  // hi | type=2
    i32x8 g1;
    g1[0] = (int)(1u << 16);                                // data_size=2B
    g1[1] = (int)((tensor_d0 & 0xffffu) << 16);             // dim0[15:0]
    g1[2] = (int)((tensor_d0 >> 16) | ((tensor_d1 & 0xffffu) << 16));
    g1[3] = (int)((tensor_d1 >> 16) | ((tile_d0 & 0xffffu) << 16));
    g1[4] = (int)(tile_d1 & 0xffffu);                       // tile_dim1, dim2=0
    g1[5] = (int)(stride0 & 0xffffffffull);                 // dim0_stride lo
    g1[6] = (int)((stride0 >> 32) & 0xffffull);             // dim0_stride hi
    g1[7] = 0;
    i32x4 z4 = {0, 0, 0, 0};
#if __clang_major__ >= 23
    i32x8 z8 = {0, 0, 0, 0, 0, 0, 0, 0};
    __builtin_amdgcn_tensor_load_to_lds(g0, g1, z4, z4, z8, 0);
#else
    __builtin_amdgcn_tensor_load_to_lds(g0, g1, z4, z4, 0);
#endif
}
#endif

// ------------------------- WMMA fragment helpers ---------------------------
// A-matrix 16x32 bf16 lane layout as loaded here (contiguous k, not the ISA table order):
//   lane<16 : row=lane,    K in {0..15}
//   lane>=16: row=lane-16, K in {16..31}
// B-matrix 32x16 bf16: lane half selects K 0..15 / 16..31, col = lane&15.
// C/D 16x16 f32: VGPR r -> M = r + 8*half, N = lane&15.

template <int PAD>
__device__ inline v16bf load_a_frag(const __bf16 (*tile)[PAD], int rowBase, int koff, int lane) {
    const int half = lane >> 4, m = lane & 15;
    v16bf a;
#pragma unroll
    for (int i = 0; i < 8; ++i) {
        const int kk = half * 16 + 2 * i + koff;
        a[2 * i]     = tile[rowBase + m][kk];
        a[2 * i + 1] = tile[rowBase + m][kk + 1];
    }
    return a;
}

template <int PAD>
__device__ inline v16bf load_b_frag_rows(const __bf16 (*tile)[PAD], int colBase, int koff, int lane) {
    // tile stored [n][k]; B(k,n) = tile[n][k]
    const int half = lane >> 4, n = lane & 15;
    v16bf b;
#pragma unroll
    for (int i = 0; i < 8; ++i) {
        const int kk = half * 16 + 2 * i + koff;
        b[2 * i]     = tile[colBase + n][kk];
        b[2 * i + 1] = tile[colBase + n][kk + 1];
    }
    return b;
}

template <int PAD>
__device__ inline v16bf load_b_frag_cols(const __bf16 (*tile)[PAD], int colBase, int koff, int lane) {
    // tile stored [k][n]; B(k,n) = tile[k][n]
    const int half = lane >> 4, n = lane & 15;
    v16bf b;
#pragma unroll
    for (int i = 0; i < 8; ++i) {
        const int kk = half * 16 + 2 * i + koff;
        b[2 * i]     = tile[kk][colBase + n];
        b[2 * i + 1] = tile[kk + 1][colBase + n];
    }
    return b;
}

#define WMMA_BF16(ACC, A, B) \
    (ACC) = __builtin_amdgcn_wmma_f32_16x16x32_bf16(false, (A), false, (B), (short)0, (ACC), false, false)

// ------------------- Kernel 1: QKV projection (x @ W^T) --------------------
// grid (NROW/128, DIM/64, 3), block 256 (8 waves as 4x2, 32x32 C per wave).
__global__ __launch_bounds__(256)
void ga_proj_kernel(const float* __restrict__ x,
                    const float* __restrict__ Wq,
                    const float* __restrict__ Wk,
                    const float* __restrict__ Wv,
                    __bf16* __restrict__ Q,
                    __bf16* __restrict__ K,
                    __bf16* __restrict__ V) {
    const float*  W = (blockIdx.z == 0) ? Wq : ((blockIdx.z == 1) ? Wk : Wv);
    __bf16*       O = (blockIdx.z == 0) ? Q  : ((blockIdx.z == 1) ? K  : V);

    __shared__ __bf16 As[128][34];  // x tile [m][k]
    __shared__ __bf16 Bs[64][34];   // W tile [n][k]

    const int t = threadIdx.x, lane = t & 31, wid = t >> 5;
    const int wm = wid & 3, wn = wid >> 2;
    const int rowBase = blockIdx.x * 128;
    const int colBase = blockIdx.y * 64;

    v8f c00 = {}; v8f c01 = {}; v8f c10 = {}; v8f c11 = {};

    for (int k0 = 0; k0 < DIM; k0 += 32) {
#pragma unroll
        for (int e = 0; e < 16; ++e) {   // A: 128x32 = 4096 elems, 16/thread
            const int idx = t * 16 + e;
            const int r = idx >> 5, c = idx & 31;
            As[r][c] = (__bf16)x[(size_t)(rowBase + r) * DIM + k0 + c];
        }
#pragma unroll
        for (int e = 0; e < 8; ++e) {    // B: 64x32 = 2048 elems, 8/thread
            const int idx = t * 8 + e;
            const int r = idx >> 5, c = idx & 31;
            Bs[r][c] = (__bf16)W[(size_t)(colBase + r) * DIM + k0 + c];
        }
        if (k0 + 32 < DIM)
            __builtin_prefetch(&x[(size_t)(rowBase + (t >> 1)) * DIM + k0 + 32], 0, 1);
        __syncthreads();

        v16bf a0 = load_a_frag<34>(As, wm * 32,      0, lane);
        v16bf a1 = load_a_frag<34>(As, wm * 32 + 16, 0, lane);
        v16bf b0 = load_b_frag_rows<34>(Bs, wn * 32,      0, lane);
        v16bf b1 = load_b_frag_rows<34>(Bs, wn * 32 + 16, 0, lane);
        WMMA_BF16(c00, a0, b0);
        WMMA_BF16(c01, a0, b1);
        WMMA_BF16(c10, a1, b0);
        WMMA_BF16(c11, a1, b1);
        __syncthreads();
    }

    const int half = lane >> 4, n = lane & 15;
#pragma unroll
    for (int r = 0; r < 8; ++r) {
        const int gm0 = rowBase + wm * 32 + r + 8 * half;
        const int gm1 = gm0 + 16;
        const int gn0 = colBase + wn * 32 + n;
        O[(size_t)gm0 * DIM + gn0]      = (__bf16)c00[r];
        O[(size_t)gm0 * DIM + gn0 + 16] = (__bf16)c01[r];
        O[(size_t)gm1 * DIM + gn0]      = (__bf16)c10[r];
        O[(size_t)gm1 * DIM + gn0 + 16] = (__bf16)c11[r];
    }
}

// --------- Kernel 2: per-row manifold scale + projected squared norm -------
__global__ __launch_bounds__(256)
void ga_norm_kernel(const __bf16* __restrict__ Q,
                    const __bf16* __restrict__ K,
                    const float* __restrict__ curv,
                    float* __restrict__ qsc, float* __restrict__ qn2,
                    float* __restrict__ ksc, float* __restrict__ kn2) {
    const int lane = threadIdx.x & 31;
    const int row  = blockIdx.x * 8 + (threadIdx.x >> 5);

    float sq = 0.f, sk = 0.f;
#pragma unroll
    for (int i = 0; i < DIM / 32; ++i) {
        const int c = lane + 32 * i;
        const float qv = (float)Q[(size_t)row * DIM + c];
        const float kv = (float)K[(size_t)row * DIM + c];
        sq += qv * qv;
        sk += kv * kv;
    }
#pragma unroll
    for (int off = 16; off > 0; off >>= 1) {
        sq += __shfl_xor(sq, off, 32);
        sk += __shfl_xor(sk, off, 32);
    }
    if (lane == 0) {
        const float kc = curv[0];
        const float cH = fmaxf(-kc, 1e-5f);
        const float mx = (1.0f - 1e-3f) * rsqrtf(cH);
        const float kp = fmaxf(kc, 1e-5f);

        float nq = sqrtf(fmaxf(sq, 1e-5f));
        float sQ = (kc < -THRC) ? fminf(1.f, mx / nq)
                 : (kc >  THRC) ? (1.f / (nq * sqrtf(kp))) : 1.f;
        qsc[row] = sQ;  qn2[row] = sq * sQ * sQ;

        float nk = sqrtf(fmaxf(sk, 1e-5f));
        float sK = (kc < -THRC) ? fminf(1.f, mx / nk)
                 : (kc >  THRC) ? (1.f / (nk * sqrtf(kp))) : 1.f;
        ksc[row] = sK;  kn2[row] = sk * sK * sK;
    }
}

// ---------------- geometric distance -> similarity epilogue ----------------
__device__ inline float geo_sim(float ip, float qn2v, float kn2v, float kc) {
    const float sqd = fmaxf(qn2v + kn2v - 2.f * ip, 0.f);
    const float euc = sqrtf(sqd + 1e-5f);
    const float cH  = fmaxf(-kc, 1e-5f);
    const float den = fmaxf((1.f - cH * qn2v) * (1.f - cH * kn2v), 1e-5f);
    const float hyp = acoshf(fmaxf(1.f + 2.f * cH * sqd / den, 1.f + 1e-7f)) * rsqrtf(cH);
    const float kp  = fmaxf(kc, 1e-5f);
    const float arg = fminf(fmaxf(kp * ip, -1.f + 1e-6f), 1.f - 1e-6f);
    const float sph = acosf(arg) * rsqrtf(kp);
    const float dist = (kc < -THRC) ? hyp : ((kc > THRC) ? sph : euc);
    return -dist * INV_SQRT_D;
}

// ---------------- Kernel 3: logits = geo(Q K^T) into attn ------------------
// grid (SEQ/128, SEQ/64, BATCH); Q/K tiles (K-step 64) staged via TDM.
__global__ __launch_bounds__(256)
void ga_logits_kernel(const __bf16* __restrict__ Q,
                      const __bf16* __restrict__ K,
                      const float* __restrict__ qsc, const float* __restrict__ qn2,
                      const float* __restrict__ ksc, const float* __restrict__ kn2,
                      const float* __restrict__ curv,
                      float* __restrict__ attn) {
    __shared__ __bf16 Qs[128][64];  // [m][k], packed (TDM writes rows contiguously)
    __shared__ __bf16 Ks[64][64];   // [n][k]

    const int t = threadIdx.x, lane = t & 31, wid = t >> 5;
    const int wm = wid & 3, wn = wid >> 2;
    const int rowBase = blockIdx.x * 128;
    const int colBase = blockIdx.y * 64;
    const int b = blockIdx.z;
    const size_t qoff = (size_t)b * SEQ + rowBase;
    const size_t koff = (size_t)b * SEQ + colBase;
#if GA_HAVE_TDM
    const unsigned qs_lds = (unsigned)(size_t)&Qs[0][0];
    const unsigned ks_lds = (unsigned)(size_t)&Ks[0][0];
#endif

    v8f c00 = {}; v8f c01 = {}; v8f c10 = {}; v8f c11 = {};

    for (int k0 = 0; k0 < DIM; k0 += 64) {
#if GA_HAVE_TDM
        if (wid == 0) {   // one descriptor per tile; TDM ignores EXEC
            tdm_load_2d(qs_lds, &Q[qoff * DIM + k0], 64, 128, DIM, NROW, DIM);
            tdm_load_2d(ks_lds, &K[koff * DIM + k0], 64, 64,  DIM, NROW, DIM);
            __builtin_amdgcn_s_wait_tensorcnt(0);
        }
        __syncthreads();
#else
#pragma unroll
        for (int e = 0; e < 32; ++e) {
            const int idx = t * 32 + e;
            Qs[idx >> 6][idx & 63] = Q[(qoff + (idx >> 6)) * DIM + k0 + (idx & 63)];
        }
#pragma unroll
        for (int e = 0; e < 16; ++e) {
            const int idx = t * 16 + e;
            Ks[idx >> 6][idx & 63] = K[(koff + (idx >> 6)) * DIM + k0 + (idx & 63)];
        }
        __syncthreads();
#endif

#pragma unroll
        for (int kk = 0; kk < 64; kk += 32) {
            v16bf a0 = load_a_frag<64>(Qs, wm * 32,      kk, lane);
            v16bf a1 = load_a_frag<64>(Qs, wm * 32 + 16, kk, lane);
            v16bf b0 = load_b_frag_rows<64>(Ks, wn * 32,      kk, lane);
            v16bf b1 = load_b_frag_rows<64>(Ks, wn * 32 + 16, kk, lane);
            WMMA_BF16(c00, a0, b0);
            WMMA_BF16(c01, a0, b1);
            WMMA_BF16(c10, a1, b0);
            WMMA_BF16(c11, a1, b1);
        }
        __syncthreads();
    }

    const float kc = curv[0];
    const int half = lane >> 4, n = lane & 15;
    const int kc0 = colBase + wn * 32 + n;
    const int kc1 = kc0 + 16;
    const float sK0 = ksc[(size_t)b * SEQ + kc0], nK0 = kn2[(size_t)b * SEQ + kc0];
    const float sK1 = ksc[(size_t)b * SEQ + kc1], nK1 = kn2[(size_t)b * SEQ + kc1];

#pragma unroll
    for (int r = 0; r < 8; ++r) {
        const int q0 = rowBase + wm * 32 + r + 8 * half;
        const int q1 = q0 + 16;
        const float sQ0 = qsc[(size_t)b * SEQ + q0], nQ0 = qn2[(size_t)b * SEQ + q0];
        const float sQ1 = qsc[(size_t)b * SEQ + q1], nQ1 = qn2[(size_t)b * SEQ + q1];
        const size_t base0 = (size_t)b * SEQ * SEQ + (size_t)q0 * SEQ;
        const size_t base1 = (size_t)b * SEQ * SEQ + (size_t)q1 * SEQ;
        attn[base0 + kc0] = geo_sim(c00[r] * sQ0 * sK0, nQ0, nK0, kc);
        attn[base0 + kc1] = geo_sim(c01[r] * sQ0 * sK1, nQ0, nK1, kc);
        attn[base1 + kc0] = geo_sim(c10[r] * sQ1 * sK0, nQ1, nK0, kc);
        attn[base1 + kc1] = geo_sim(c11[r] * sQ1 * sK1, nQ1, nK1, kc);
    }
}

// --------------------- Kernel 4: in-place row softmax ----------------------
__global__ __launch_bounds__(256)
void ga_softmax_kernel(float* __restrict__ attn) {
    __shared__ float red[256];
    const int t = threadIdx.x;
    const size_t base = (size_t)blockIdx.x * SEQ;

    float v[16];
    float mx = -3.4e38f;
#pragma unroll
    for (int i = 0; i < 16; ++i) {
        v[i] = attn[base + t + 256 * i];
        mx = fmaxf(mx, v[i]);
    }
    red[t] = mx; __syncthreads();
#pragma unroll
    for (int s = 128; s > 0; s >>= 1) {
        if (t < s) red[t] = fmaxf(red[t], red[t + s]);
        __syncthreads();
    }
    mx = red[0]; __syncthreads();

    float sum = 0.f;
#pragma unroll
    for (int i = 0; i < 16; ++i) { v[i] = __expf(v[i] - mx); sum += v[i]; }
    red[t] = sum; __syncthreads();
#pragma unroll
    for (int s = 128; s > 0; s >>= 1) {
        if (t < s) red[t] += red[t + s];
        __syncthreads();
    }
    const float inv = 1.f / red[0];
#pragma unroll
    for (int i = 0; i < 16; ++i) attn[base + t + 256 * i] = v[i] * inv;
}

// ---------------------- Kernel 5: out = attn @ V ---------------------------
// grid (SEQ/128, DIM/64, BATCH); V tiles (K-step 64) via TDM, attn converted.
__global__ __launch_bounds__(256)
void ga_outgemm_kernel(const float* __restrict__ attn,
                       const __bf16* __restrict__ V,
                       float* __restrict__ out) {
    __shared__ __bf16 As[128][64];  // attn tile [m][k] (fp32 -> bf16)
    __shared__ __bf16 Vs[64][64];   // V tile    [k][n] (TDM, packed)

    const int t = threadIdx.x, lane = t & 31, wid = t >> 5;
    const int wm = wid & 3, wn = wid >> 2;
    const int rowBase = blockIdx.x * 128;
    const int colBase = blockIdx.y * 64;
    const int b = blockIdx.z;
    const size_t abase = (size_t)b * SEQ * SEQ;
#if GA_HAVE_TDM
    const unsigned vs_lds = (unsigned)(size_t)&Vs[0][0];
#endif

    v8f c00 = {}; v8f c01 = {}; v8f c10 = {}; v8f c11 = {};

    for (int k0 = 0; k0 < SEQ; k0 += 64) {
#if GA_HAVE_TDM
        if (wid == 0) {
            tdm_load_2d(vs_lds, &V[((size_t)b * SEQ + k0) * DIM + colBase],
                        64, 64, DIM, NROW, DIM);
            __builtin_amdgcn_s_wait_tensorcnt(0);
        }
#else
#pragma unroll
        for (int e = 0; e < 16; ++e) {
            const int idx = t * 16 + e;
            Vs[idx >> 6][idx & 63] =
                V[((size_t)b * SEQ + k0 + (idx >> 6)) * DIM + colBase + (idx & 63)];
        }
#endif
#pragma unroll
        for (int e = 0; e < 32; ++e) {   // attn: 128x64 fp32 -> bf16
            const int idx = t * 32 + e;
            const int r = idx >> 6, c = idx & 63;
            As[r][c] = (__bf16)attn[abase + (size_t)(rowBase + r) * SEQ + k0 + c];
        }
        if (k0 + 64 < SEQ)
            __builtin_prefetch(&attn[abase + (size_t)(rowBase + (t >> 1)) * SEQ + k0 + 64], 0, 1);
        __syncthreads();

#pragma unroll
        for (int kk = 0; kk < 64; kk += 32) {
            v16bf a0 = load_a_frag<64>(As, wm * 32,      kk, lane);
            v16bf a1 = load_a_frag<64>(As, wm * 32 + 16, kk, lane);
            v16bf b0 = load_b_frag_cols<64>(Vs, wn * 32,      kk, lane);
            v16bf b1 = load_b_frag_cols<64>(Vs, wn * 32 + 16, kk, lane);
            WMMA_BF16(c00, a0, b0);
            WMMA_BF16(c01, a0, b1);
            WMMA_BF16(c10, a1, b0);
            WMMA_BF16(c11, a1, b1);
        }
        __syncthreads();
    }

    const int half = lane >> 4, n = lane & 15;
#pragma unroll
    for (int r = 0; r < 8; ++r) {
        const size_t gm0 = (size_t)b * SEQ + rowBase + wm * 32 + r + 8 * half;
        const size_t gm1 = gm0 + 16;
        const int gn0 = colBase + wn * 32 + n;
        out[gm0 * DIM + gn0]      = c00[r];
        out[gm0 * DIM + gn0 + 16] = c01[r];
        out[gm1 * DIM + gn0]      = c10[r];
        out[gm1 * DIM + gn0 + 16] = c11[r];
    }
}

// ------------------------ Kernel 6: write k scalar -------------------------
__global__ void ga_scalar_kernel(const float* __restrict__ curv, float* __restrict__ dst) {
    if (threadIdx.x == 0 && blockIdx.x == 0) dst[0] = curv[0];
}

// ---------------------------------------------------------------------------
extern "C" void kernel_launch(void* const* d_in, const int* in_sizes, int n_in,
                              void* d_out, int out_size, void* d_ws, size_t ws_size,
                              hipStream_t stream) {
    const float* x    = (const float*)d_in[0];
    const float* Wq   = (const float*)d_in[1];
    const float* Wk   = (const float*)d_in[2];
    const float* Wv   = (const float*)d_in[3];
    const float* curv = (const float*)d_in[4];

    float* out  = (float*)d_out;                       // [B,S,D]
    float* attn = out + (size_t)BATCH * SEQ * DIM;     // [B,S,S]
    float* kout = attn + (size_t)BATCH * SEQ * SEQ;    // [1]

    // Workspace: Q,K,V bf16 (16 MB each) + 4 per-row float arrays (~256 KB).
    char* ws = (char*)d_ws;
    __bf16* Q = (__bf16*)ws;
    __bf16* K = Q + (size_t)NROW * DIM;
    __bf16* V = K + (size_t)NROW * DIM;
    float* qsc = (float*)(V + (size_t)NROW * DIM);
    float* qn2 = qsc + NROW;
    float* ksc = qn2 + NROW;
    float* kn2 = ksc + NROW;

    ga_proj_kernel<<<dim3(NROW / 128, DIM / 64, 3), 256, 0, stream>>>(x, Wq, Wk, Wv, Q, K, V);
    ga_norm_kernel<<<dim3(NROW / 8), 256, 0, stream>>>(Q, K, curv, qsc, qn2, ksc, kn2);
    ga_logits_kernel<<<dim3(SEQ / 128, SEQ / 64, BATCH), 256, 0, stream>>>(Q, K, qsc, qn2, ksc, kn2, curv, attn);
    ga_softmax_kernel<<<dim3(NROW), 256, 0, stream>>>(attn);
    ga_outgemm_kernel<<<dim3(SEQ / 128, DIM / 64, BATCH), 256, 0, stream>>>(attn, V, out);
    ga_scalar_kernel<<<1, 32, 0, stream>>>(curv, kout);
}
